// AttentionNeck1_12077448036704
// MI455X (gfx1250) — matrix operand order repaired and checked by reading, not compiled, not run on hardware
//
#include <hip/hip_runtime.h>
#include <hip/hip_bf16.h>
#include <math.h>

// ---------------------------------------------------------------------------
// Problem constants (reference: B=16, IN_C=1024, W=H=32 -> N=1024, P=256,
// HEADS=4, D=64)
// ---------------------------------------------------------------------------
static constexpr int BATCH = 16;
static constexpr int INC   = 1024;
static constexpr int PCH   = 256;
static constexpr int HEADS = 4;
static constexpr int DHEAD = 64;
static constexpr int NTOK  = 1024;   // W*H
static constexpr float LN_EPS = 1e-6f;

#define USE_ASYNC_LDS 1

typedef _Float16 f16_t;
typedef f16_t v16h __attribute__((ext_vector_type(16)));
typedef f16_t v8h  __attribute__((ext_vector_type(8)));
typedef float v8f  __attribute__((ext_vector_type(8)));

#define DEVINL static __device__ __forceinline__

DEVINL v16h ld_v16(const f16_t* p) {
  v8h a = *(const v8h*)p;
  v8h b = *(const v8h*)(p + 8);
  return __builtin_shufflevector(a, b, 0,1,2,3,4,5,6,7,8,9,10,11,12,13,14,15);
}
DEVINL v16h ld_v8x2(const f16_t* p0, const f16_t* p1) {
  v8h a = *(const v8h*)p0;
  v8h b = *(const v8h*)p1;
  return __builtin_shufflevector(a, b, 0,1,2,3,4,5,6,7,8,9,10,11,12,13,14,15);
}
// D = A(16x32 f16) * B(32x16 f16) + C(16x16 f32)
DEVINL v8f wmma16(v16h a, v16h b, v8f c) {
  return __builtin_amdgcn_wmma_f32_16x16x32_f16(
      /*neg_a=*/false, a, /*neg_b=*/false, b,
      /*c_mod=*/(short)0, c, /*reuse_a=*/false, /*reuse_b=*/false);
}

#if USE_ASYNC_LDS
// Async DMA of 16B per lane: global -> LDS, tracked on ASYNCcnt.
// Low 32 bits of a generic pointer to __shared__ are the LDS byte address
// (ISA 10.2: LDS aperture truncates to addr[31:0]).
DEVINL void async_ld_b128(void* lds, const void* gaddr) {
  unsigned int l = (unsigned int)(unsigned long long)lds;
  asm volatile("global_load_async_to_lds_b128 %0, %1, off"
               :: "v"(l), "v"(gaddr) : "memory");
}
DEVINL void wait_async0() { asm volatile("s_wait_asynccnt 0" ::: "memory"); }
#endif

// Max-reduce across the 16 lanes of a DPP row via rotate-reduce (VALU-only,
// co-executes with XDL WMMA; keeps the DS pipe free for fragment loads).
#if __has_builtin(__builtin_amdgcn_update_dpp)
#define ROWRED_STEP(vf, ctrl)                                                 \
  do {                                                                        \
    int _u = __float_as_int(vf);                                              \
    int _p = __builtin_amdgcn_update_dpp(_u, _u, (ctrl), 0xf, 0xf, true);     \
    vf = fmaxf(vf, __int_as_float(_p));                                       \
  } while (0)
DEVINL float rowmax16(float v) {
  ROWRED_STEP(v, 0x128);   // row_ror:8
  ROWRED_STEP(v, 0x124);   // row_ror:4
  ROWRED_STEP(v, 0x122);   // row_ror:2
  ROWRED_STEP(v, 0x121);   // row_ror:1
  return v;
}
#else
DEVINL float rowmax16(float v) {
#pragma unroll
  for (int msk = 8; msk >= 1; msk >>= 1)
    v = fmaxf(v, __shfl_xor(v, msk, 32));
  return v;
}
#endif

// ---------------------------------------------------------------------------
// Prep kernels
// ---------------------------------------------------------------------------
__global__ void cvt_f32_f16_kernel(const float* __restrict__ in,
                                   f16_t* __restrict__ out, int n) {
  for (int i = blockIdx.x * blockDim.x + threadIdx.x; i < n;
       i += gridDim.x * blockDim.x)
    out[i] = (f16_t)in[i];
}

// post[h][n][d] = rel_h[h][d][n%32] + rel_w[h][d][n/32]   (n = w*32 + hh)
__global__ void pos_kernel(const float* __restrict__ rel_h,
                           const float* __restrict__ rel_w,
                           f16_t* __restrict__ post) {
  int t = blockIdx.x * blockDim.x + threadIdx.x;      // HEADS*NTOK*DHEAD
  if (t >= HEADS * NTOK * DHEAD) return;
  int d = t & 63;
  int n = (t >> 6) & (NTOK - 1);
  int h = t >> 16;
  int hh = n & 31, ww = n >> 5;
  float v = rel_h[(h * DHEAD + d) * 32 + hh] + rel_w[(h * DHEAD + d) * 32 + ww];
  post[t] = (f16_t)v;
}

// x[b][c][n] f32  ->  xt[b][n][c] f16   (32x32 LDS tile transpose)
__global__ __launch_bounds__(256) void transpose_x_kernel(
    const float* __restrict__ x, f16_t* __restrict__ xt) {
  __shared__ float tile[32][33];
  int b  = blockIdx.z;
  int cb = blockIdx.y * 32;
  int nb = blockIdx.x * 32;
  int tx = threadIdx.x & 31, ty = threadIdx.x >> 5;   // 32 x 8
#pragma unroll
  for (int i = 0; i < 32; i += 8)
    tile[ty + i][tx] = x[((size_t)b * INC + cb + ty + i) * NTOK + nb + tx];
  __syncthreads();
#pragma unroll
  for (int i = 0; i < 32; i += 8)
    xt[((size_t)b * NTOK + nb + ty + i) * INC + cb + tx] = (f16_t)tile[tx][ty + i];
}

// ---------------------------------------------------------------------------
// Generic WMMA GEMM:  C[b] = A * B[b],  A: [M,K] f16 row-major,
// Bt: [b][N][K] f16 (B transposed, K contiguous).  Tile 128x64, BK=64,
// double-buffered LDS fed by async global->LDS DMA, 8 waves (4x2),
// each wave 32x32 = 2x2 fragments.
// MODE 0: store outT[b][n][M] f16                         (down conv)
// MODE 1: +bias, store qt[b][h][n][d] f16 (p = h*64+d)    (q / k proj)
// MODE 2: +bias, store v [b][h][d][n] f16                 (v proj)
// MODE 3: gelu(C) + xres, store f32 [b][M][n]             (expand conv)
// ---------------------------------------------------------------------------
template <int MODE>
__global__ __launch_bounds__(256, 1) void gemm_kernel(
    const f16_t* __restrict__ A, const f16_t* __restrict__ Bt,
    const float* __restrict__ bias, f16_t* __restrict__ outH,
    float* __restrict__ outF, const float* __restrict__ xres,
    int M, int N, int K) {
  __shared__ __align__(16) f16_t As[2][128][80];
  __shared__ __align__(16) f16_t Bs[2][64][80];

  const int b      = blockIdx.y;
  const int tilesN = N >> 6;
  const int mTile  = (blockIdx.x / tilesN) * 128;
  const int nTile  = (blockIdx.x % tilesN) * 64;
  const f16_t* Bb  = Bt + (size_t)b * N * K;

  const int t    = threadIdx.x;
  const int lane = t & 31;
  const int w    = t >> 5;
  const int wm   = w >> 1, wn = w & 1;
  const int r    = lane & 15, hi = lane >> 4;
  const int rowA = t >> 3, colA = (t & 7) * 8;

#if USE_ASYNC_LDS
  auto stage = [&](int bb, int k0) {
#pragma unroll
    for (int i = 0; i < 4; ++i) {
      int ci = t + i * 256;
      int row = ci >> 3, c8 = (ci & 7) * 8;
      async_ld_b128(&As[bb][row][c8], &A[(size_t)(mTile + row) * K + k0 + c8]);
    }
#pragma unroll
    for (int i = 0; i < 2; ++i) {
      int ci = t + i * 256;
      int row = ci >> 3, c8 = (ci & 7) * 8;
      async_ld_b128(&Bs[bb][row][c8], &Bb[(size_t)(nTile + row) * K + k0 + c8]);
    }
  };
#else
  v8h aR[4], bR[2];
  auto stage_ld = [&](int k0) {
#pragma unroll
    for (int i = 0; i < 4; ++i) {
      int ci = t + i * 256;
      int row = ci >> 3, c8 = (ci & 7) * 8;
      aR[i] = *(const v8h*)&A[(size_t)(mTile + row) * K + k0 + c8];
    }
#pragma unroll
    for (int i = 0; i < 2; ++i) {
      int ci = t + i * 256;
      int row = ci >> 3, c8 = (ci & 7) * 8;
      bR[i] = *(const v8h*)&Bb[(size_t)(nTile + row) * K + k0 + c8];
    }
  };
  auto stage_st = [&](int bb) {
#pragma unroll
    for (int i = 0; i < 4; ++i) {
      int ci = t + i * 256;
      int row = ci >> 3, c8 = (ci & 7) * 8;
      *(v8h*)&As[bb][row][c8] = aR[i];
    }
#pragma unroll
    for (int i = 0; i < 2; ++i) {
      int ci = t + i * 256;
      int row = ci >> 3, c8 = (ci & 7) * 8;
      *(v8h*)&Bs[bb][row][c8] = bR[i];
    }
  };
#endif

  v8f acc[2][2] = {};

#if USE_ASYNC_LDS
  stage(0, 0);
  wait_async0();
#else
  stage_ld(0);
  stage_st(0);
#endif
  __syncthreads();

  const int nT = K >> 6;
  for (int it = 0; it < nT; ++it) {
    const int cur = it & 1;
#if USE_ASYNC_LDS
    if (it + 1 < nT) stage(cur ^ 1, (it + 1) * 64);   // DMA in flight
#else
    if (it + 1 < nT) stage_ld((it + 1) * 64);
#endif
    if (it + 2 < nT) {                                // prefetch tile after next
      __builtin_prefetch(&A[(size_t)(mTile + rowA) * K + (it + 2) * 64 + colA], 0, 1);
      __builtin_prefetch(&Bb[(size_t)(nTile + (t & 63)) * K + (it + 2) * 64], 0, 1);
    }
#pragma unroll
    for (int kk = 0; kk < 64; kk += 32) {
      v16h aF[2], bF[2];
#pragma unroll
      for (int mi = 0; mi < 2; ++mi)
        aF[mi] = ld_v16(&As[cur][wm * 32 + mi * 16 + r][kk + hi * 16]);
#pragma unroll
      for (int ni = 0; ni < 2; ++ni)
        bF[ni] = ld_v16(&Bs[cur][wn * 32 + ni * 16 + r][kk + hi * 16]);
#pragma unroll
      for (int mi = 0; mi < 2; ++mi)
#pragma unroll
        for (int ni = 0; ni < 2; ++ni)
          acc[mi][ni] = wmma16(aF[mi], bF[ni], acc[mi][ni]);
    }
#if USE_ASYNC_LDS
    wait_async0();
#else
    if (it + 1 < nT) stage_st(cur ^ 1);
#endif
    __syncthreads();
  }

  // epilogue: C element (vr,lane) -> M = vr + 8*hi, N = r
#pragma unroll
  for (int mi = 0; mi < 2; ++mi)
#pragma unroll
    for (int ni = 0; ni < 2; ++ni)
#pragma unroll
      for (int vr = 0; vr < 8; ++vr) {
        int Mg = mTile + wm * 32 + mi * 16 + vr + 8 * hi;
        int Ng = nTile + wn * 32 + ni * 16 + r;
        float val = acc[mi][ni][vr];
        if (MODE == 0) {
          outH[((size_t)b * N + Ng) * M + Mg] = (f16_t)val;
        } else if (MODE == 1) {
          val += bias[Mg];
          int hh = Mg >> 6, dd = Mg & 63;
          outH[(((size_t)b * HEADS + hh) * NTOK + Ng) * DHEAD + dd] = (f16_t)val;
        } else if (MODE == 2) {
          val += bias[Mg];
          int hh = Mg >> 6, dd = Mg & 63;
          outH[(((size_t)b * HEADS + hh) * DHEAD + dd) * NTOK + Ng] = (f16_t)val;
        } else {
          float g = 0.5f * val * (1.0f + erff(val * 0.70710678118654752f));
          size_t idx = ((size_t)b * INC + Mg) * (size_t)NTOK + Ng;
          outF[idx] = g + xres[idx];
        }
      }
}

// ---------------------------------------------------------------------------
// Flash attention: logits[i,j] = q_i.k_j + pos_i.q_j, softmax over j,
// O[d,i] = sum_j v[d,j] * P[i,j].  One block per (b,h, 128-row i block);
// each of 8 waves owns 16 rows.  K^T/Q^T/V j-tiles staged via async DMA,
// double-buffered.  Row-sums come free from an all-ones "V" fragment (5th
// O accumulator); the row-max uses a VALU-only DPP rotate-reduce.
// qt/kt: [BH][N][64] f16, v: [BH][64][N] f16, post: [H][N][64] f16.
// ---------------------------------------------------------------------------
__global__ __launch_bounds__(256, 1) void attn_kernel(
    const f16_t* __restrict__ qt, const f16_t* __restrict__ kt,
    const f16_t* __restrict__ vv, const f16_t* __restrict__ post,
    float* __restrict__ obuf) {
  __shared__ __align__(16) f16_t sK[2][32][80];  // [j][d], padded rows
  __shared__ __align__(16) f16_t sQ[2][32][80];  // [j][d]
  __shared__ __align__(16) f16_t sV[2][64][40];  // [d][j], padded rows
  __shared__ __align__(16) f16_t sP[8][16][32];  // per-wave P tile [i][j]
  __shared__ float sA[8][16];                    // per-wave alpha broadcast

  const int t    = threadIdx.x;
  const int w    = t >> 5;
  const int lane = t & 31;
  const int r    = lane & 15, hi = lane >> 4;

  const int bh = blockIdx.x >> 3;   // 0..63
  const int ib = blockIdx.x & 7;
  const int b  = bh >> 2, h = bh & 3;
  const int i0 = ib * 128 + w * 16; // this wave's query-row base

  const f16_t* qtb = qt + (size_t)bh * NTOK * DHEAD;
  const f16_t* ktb = kt + (size_t)bh * NTOK * DHEAD;
  const f16_t* vb  = vv + (size_t)bh * DHEAD * NTOK;
  const f16_t* pb  = post + (size_t)h * NTOK * DHEAD;

  // cooperative staging: 256 chunks each for sK/sQ (32 rows x 8), sV (64 x 4)
  const int rk = t >> 3, ck = (t & 7) * 8;
  const int rv = t >> 2, cv = (t & 3) * 8;
#if USE_ASYNC_LDS
  auto stage = [&](int bb, int jb) {
    async_ld_b128(&sK[bb][rk][ck], &ktb[(size_t)(jb + rk) * DHEAD + ck]);
    async_ld_b128(&sQ[bb][rk][ck], &qtb[(size_t)(jb + rk) * DHEAD + ck]);
    async_ld_b128(&sV[bb][rv][cv], &vb[(size_t)rv * NTOK + jb + cv]);
  };
#else
  v8h rK, rQ, rV;
  auto stage_ld = [&](int jb) {
    rK = *(const v8h*)&ktb[(size_t)(jb + rk) * DHEAD + ck];
    rQ = *(const v8h*)&qtb[(size_t)(jb + rk) * DHEAD + ck];
    rV = *(const v8h*)&vb[(size_t)rv * NTOK + jb + cv];
  };
  auto stage_st = [&](int bb) {
    *(v8h*)&sK[bb][rk][ck] = rK;
    *(v8h*)&sQ[bb][rk][ck] = rQ;
    *(v8h*)&sV[bb][rv][cv] = rV;
  };
#endif

  // persistent A-fragments: Q^T and Pos^T rows [i0..i0+16) x K=64 (2 frags)
  v16h qA[2], pA[2];
#pragma unroll
  for (int f = 0; f < 2; ++f) {
    const f16_t* qs = qtb + (size_t)(i0 + r) * DHEAD + f * 32 + hi * 16;
    qA[f] = ld_v8x2(qs, qs + 8);
    const f16_t* ps = pb + (size_t)(i0 + r) * DHEAD + f * 32 + hi * 16;
    pA[f] = ld_v8x2(ps, ps + 8);
  }
  v16h onesA;
#pragma unroll
  for (int i = 0; i < 16; ++i) onesA[i] = (f16_t)1.0f;

  v8f O[5] = {};                 // O[4] accumulates the row sums (ones trick)
  float mrow[8];
#pragma unroll
  for (int vr = 0; vr < 8; ++vr) mrow[vr] = -1e30f;

#if USE_ASYNC_LDS
  stage(0, 0);
  wait_async0();
#else
  stage_ld(0);
  stage_st(0);
#endif
  __syncthreads();

  for (int jb = 0; jb < NTOK; jb += 32) {
    const int cur = (jb >> 5) & 1;
#if USE_ASYNC_LDS
    if (jb + 32 < NTOK) stage(cur ^ 1, jb + 32);   // DMA next tile in flight
#else
    if (jb + 32 < NTOK) stage_ld(jb + 32);
#endif

    // ---- load ALL S-phase fragments first, then the WMMA chain ----
    v16h kB[2][2], qB[2][2];
#pragma unroll
    for (int s = 0; s < 2; ++s)
#pragma unroll
      for (int f = 0; f < 2; ++f) {
        kB[s][f] = ld_v16(&sK[cur][s * 16 + r][f * 32 + hi * 16]);
        qB[s][f] = ld_v16(&sQ[cur][s * 16 + r][f * 32 + hi * 16]);
      }
    v8f S[2] = {};
#pragma unroll
    for (int s = 0; s < 2; ++s)
#pragma unroll
      for (int f = 0; f < 2; ++f) {
        S[s] = wmma16(qA[f], kB[s][f], S[s]);
        S[s] = wmma16(pA[f], qB[s][f], S[s]);
      }

    // ---- online softmax: row-max via VALU-only DPP rotate-reduce ----
    float alpha[8];
#pragma unroll
    for (int vr = 0; vr < 8; ++vr) {
      float cmax = rowmax16(fmaxf(S[0][vr], S[1][vr]));
      float mnew = fmaxf(mrow[vr], cmax);
      alpha[vr]  = __expf(mrow[vr] - mnew);
      mrow[vr]   = mnew;
      sP[w][vr + 8 * hi][r]      = (f16_t)__expf(S[0][vr] - mnew);
      sP[w][vr + 8 * hi][r + 16] = (f16_t)__expf(S[1][vr] - mnew);
    }
    if (r == 0) {
#pragma unroll
      for (int vr = 0; vr < 8; ++vr) sA[w][8 * hi + vr] = alpha[vr];
    }
    // cross-lane LDS dep within the wave; DS ops are in-order per wave.
    asm volatile("s_wait_dscnt 0" ::: "memory");

    float av = sA[w][r];                      // alpha for column i = r
#pragma unroll
    for (int f = 0; f < 5; ++f)
#pragma unroll
      for (int vr = 0; vr < 8; ++vr) O[f][vr] *= av;

    v16h pB = ld_v16(&sP[w][r][hi * 16]);     // P^T B-fragment (K=32 j)
    v16h vA[4];
#pragma unroll
    for (int f = 0; f < 4; ++f)
      vA[f] = ld_v16(&sV[cur][f * 16 + r][hi * 16]);
#pragma unroll
    for (int f = 0; f < 4; ++f) O[f] = wmma16(vA[f], pB, O[f]);
    O[4] = wmma16(onesA, pB, O[4]);           // running row-sum l[i]

#if USE_ASYNC_LDS
    wait_async0();
#else
    if (jb + 32 < NTOK) stage_st(cur ^ 1);
#endif
    __syncthreads();
  }

  // ---- normalize and store: o[b][h*64+d][i] f32 ----
  const float inv = 1.0f / O[4][0];           // l for column i = r (all rows equal)
  const int n = i0 + r;
#pragma unroll
  for (int f = 0; f < 4; ++f)
#pragma unroll
    for (int vr = 0; vr < 8; ++vr) {
      int d = f * 16 + vr + 8 * hi;
      obuf[((size_t)b * PCH + h * DHEAD + d) * NTOK + n] = O[f][vr] * inv;
    }
}

// ---------------------------------------------------------------------------
// LayerNorm over channels per (b,n); writes y transposed: ybt[b][n][p] f16
// ---------------------------------------------------------------------------
__global__ __launch_bounds__(256) void ln_kernel(
    const float* __restrict__ obuf, const float* __restrict__ ln_w,
    const float* __restrict__ ln_b, f16_t* __restrict__ ybt) {
  int t = blockIdx.x * blockDim.x + threadIdx.x;  // BATCH*NTOK
  int b = t >> 10, n = t & (NTOK - 1);
  const float* op = obuf + (size_t)b * PCH * NTOK + n;
  float s = 0.0f, ss = 0.0f;
#pragma unroll 4
  for (int p = 0; p < PCH; ++p) {
    float v = op[(size_t)p * NTOK];
    s += v; ss += v * v;
  }
  float mu  = s * (1.0f / PCH);
  float var = ss * (1.0f / PCH) - mu * mu;
  float rstd = rsqrtf(var + LN_EPS);
  f16_t* yp = ybt + ((size_t)b * NTOK + n) * PCH;
#pragma unroll 4
  for (int p = 0; p < PCH; ++p) {
    float v = (op[(size_t)p * NTOK] - mu) * rstd;
    yp[p] = (f16_t)(v * ln_w[p] + ln_b[p]);
  }
}

// ---------------------------------------------------------------------------
// Launch
// ---------------------------------------------------------------------------
extern "C" void kernel_launch(void* const* d_in, const int* in_sizes, int n_in,
                              void* d_out, int out_size, void* d_ws,
                              size_t ws_size, hipStream_t stream) {
  (void)in_sizes; (void)n_in; (void)out_size; (void)ws_size;

  const float* x     = (const float*)d_in[0];
  const float* w_ds  = (const float*)d_in[1];
  const float* wq    = (const float*)d_in[2];
  const float* bq    = (const float*)d_in[3];
  const float* wk    = (const float*)d_in[4];
  const float* bk    = (const float*)d_in[5];
  const float* wv    = (const float*)d_in[6];
  const float* bv    = (const float*)d_in[7];
  const float* rel_h = (const float*)d_in[8];
  const float* rel_w = (const float*)d_in[9];
  const float* ln_w  = (const float*)d_in[10];
  const float* ln_b  = (const float*)d_in[11];
  const float* w1    = (const float*)d_in[12];
  float* out = (float*)d_out;

  char* ws = (char*)d_ws;
  size_t off = 0;
  auto carve = [&](size_t bytes) {
    void* p = ws + off;
    off += (bytes + 255) & ~(size_t)255;
    return p;
  };
  f16_t* xt   = (f16_t*)carve((size_t)BATCH * NTOK * INC * 2);       // 32MB
  f16_t* wdsh = (f16_t*)carve((size_t)PCH * INC * 2);
  f16_t* wqh  = (f16_t*)carve((size_t)PCH * PCH * 2);
  f16_t* wkh  = (f16_t*)carve((size_t)PCH * PCH * 2);
  f16_t* wvh  = (f16_t*)carve((size_t)PCH * PCH * 2);
  f16_t* w1h  = (f16_t*)carve((size_t)INC * PCH * 2);
  f16_t* post = (f16_t*)carve((size_t)HEADS * NTOK * DHEAD * 2);
  f16_t* outT = (f16_t*)carve((size_t)BATCH * NTOK * PCH * 2);       // 8MB
  f16_t* qT   = (f16_t*)carve((size_t)BATCH * HEADS * NTOK * DHEAD * 2);
  f16_t* kT   = (f16_t*)carve((size_t)BATCH * HEADS * NTOK * DHEAD * 2);
  f16_t* vN   = (f16_t*)carve((size_t)BATCH * HEADS * DHEAD * NTOK * 2);
  float* obuf = (float*)carve((size_t)BATCH * PCH * NTOK * 4);       // 16MB
  f16_t* ybt  = (f16_t*)carve((size_t)BATCH * NTOK * PCH * 2);       // 8MB

  // -- prep --
  cvt_f32_f16_kernel<<<512, 256, 0, stream>>>(w_ds, wdsh, PCH * INC);
  cvt_f32_f16_kernel<<<256, 256, 0, stream>>>(wq, wqh, PCH * PCH);
  cvt_f32_f16_kernel<<<256, 256, 0, stream>>>(wk, wkh, PCH * PCH);
  cvt_f32_f16_kernel<<<256, 256, 0, stream>>>(wv, wvh, PCH * PCH);
  cvt_f32_f16_kernel<<<512, 256, 0, stream>>>(w1, w1h, INC * PCH);
  pos_kernel<<<(HEADS * NTOK * DHEAD) / 256, 256, 0, stream>>>(rel_h, rel_w, post);
  transpose_x_kernel<<<dim3(NTOK / 32, INC / 32, BATCH), 256, 0, stream>>>(x, xt);

  // -- down conv: out^T = (w_ds * x)^T --
  gemm_kernel<0><<<dim3((PCH / 128) * (NTOK / 64), BATCH), 256, 0, stream>>>(
      wdsh, xt, nullptr, outT, nullptr, nullptr, PCH, NTOK, INC);

  // -- q / k / v projections --
  gemm_kernel<1><<<dim3((PCH / 128) * (NTOK / 64), BATCH), 256, 0, stream>>>(
      wqh, outT, bq, qT, nullptr, nullptr, PCH, NTOK, PCH);
  gemm_kernel<1><<<dim3((PCH / 128) * (NTOK / 64), BATCH), 256, 0, stream>>>(
      wkh, outT, bk, kT, nullptr, nullptr, PCH, NTOK, PCH);
  gemm_kernel<2><<<dim3((PCH / 128) * (NTOK / 64), BATCH), 256, 0, stream>>>(
      wvh, outT, bv, vN, nullptr, nullptr, PCH, NTOK, PCH);

  // -- flash attention --
  attn_kernel<<<BATCH * HEADS * (NTOK / 128), 256, 0, stream>>>(qT, kT, vN, post, obuf);

  // -- layernorm (writes y^T) --
  ln_kernel<<<(BATCH * NTOK) / 256, 256, 0, stream>>>(obuf, ln_w, ln_b, ybt);

  // -- expand conv + exact GELU + residual --
  gemm_kernel<3><<<dim3((INC / 128) * (NTOK / 64), BATCH), 256, 0, stream>>>(
      w1h, ybt, nullptr, nullptr, out, x, INC, NTOK, PCH);
}
